// GPTAttention_30631706755256
// MI455X (gfx1250) — matrix operand order repaired and checked by reading, not compiled, not run on hardware
//
#include <hip/hip_runtime.h>
#include <hip/hip_bf16.h>

// Problem constants (from the reference)
#define NHEAD 32
#define HDIM  64
#define HID   2048
#define QLEN  2048
#define ATT_SCALE 0.125f   // 64^-0.5

typedef __attribute__((ext_vector_type(16))) __bf16 v16bf;
typedef __attribute__((ext_vector_type(8)))  float  v8f;
typedef __attribute__((ext_vector_type(4)))  unsigned int u32x4;

union FragBF { v16bf v; u32x4 u[2]; };

#define WMMA_BF16(A, B, C) \
  __builtin_amdgcn_wmma_f32_16x16x32_bf16(false, (A), false, (B), (short)0, (C), false, false)

// ---------------------------------------------------------------------------
// A-matrix 16x32 bf16 fragment from row-major [row][k] data (stride ld).
// Lanes 0-15: row m0+lane, K = [k0..k0+7] and [k0+16..k0+23]
// Lanes 16-31: row m0+lane-16, K = [k0+8..k0+15] and [k0+24..k0+31]
__device__ __forceinline__ v16bf load_a_frag(const __bf16* A, int ld, int m0,
                                             int k0, int lane) {
  FragBF f;
  int row = m0 + (lane & 15);
  const __bf16* p = A + (size_t)row * ld + k0 + ((lane >> 4) << 3);
  f.u[0] = *(const u32x4*)(p);
  f.u[1] = *(const u32x4*)(p + 16);
  return f.v;
}

// B-matrix 32x16 bf16 fragment where B[k][n] = W[n][k] with W row-major
// (stride ld).  Lane holds column n0+(lane&15); same K order as A:
// lanes 0-15: K = [k0..k0+7],[k0+16..k0+23]; lanes 16-31: [k0+8..k0+15],[k0+24..k0+31].
__device__ __forceinline__ v16bf load_b_frag(const __bf16* W, int ld, int n0,
                                             int k0, int lane) {
  FragBF f;
  int col = n0 + (lane & 15);
  const __bf16* p = W + (size_t)col * ld + k0 + ((lane >> 4) << 3);
  f.u[0] = *(const u32x4*)(p);
  f.u[1] = *(const u32x4*)(p + 16);
  return f.v;
}

// gfx1250 async global->LDS copy, 16 bytes per lane (ASYNCcnt-tracked).
__device__ __forceinline__ void async_copy_b128(unsigned lds_off,
                                                const __bf16* gptr) {
  asm volatile("global_load_async_to_lds_b128 %0, %1, off"
               :: "v"(lds_off), "v"(gptr) : "memory");
}

// ---------------------------------------------------------------------------
__global__ void cvt_f32_bf16(const float* __restrict__ s,
                             __bf16* __restrict__ d, int n) {
  int i = blockIdx.x * blockDim.x + threadIdx.x;
  int stride = gridDim.x * blockDim.x;
  for (; i < n; i += stride) d[i] = (__bf16)s[i];
}

// ---------------------------------------------------------------------------
// Y = A(MxK,bf16) @ W(NxK,bf16)^T + bias.
// Block: 128 threads (4 waves), 128x128 block tile, 64x64 wave tile.
// K staged 32 at a time through double-buffered LDS via async copies.
// mode 0: store bf16 scaled to per-head layout [h][m][d]      (Q / K)
// mode 2: store bf16 to transposed per-head layout [h][d][m]  (V)
// mode 3: store f32 row-major [m][n]                          (final out)
#define LDS_LDK 40   // 32 K elements + 8 pad -> 80B row stride, bank-conflict-free
__global__ __launch_bounds__(128) void gemm_wmma(
    const __bf16* __restrict__ A, const __bf16* __restrict__ W,
    const float* __restrict__ bias, void* __restrict__ outp,
    int mode, float scale, int M, int N, int K) {
  __shared__ __align__(16) __bf16 sA[2][128 * LDS_LDK];
  __shared__ __align__(16) __bf16 sB[2][128 * LDS_LDK];

  int t     = threadIdx.x;
  int lane  = t & 31;
  int wave  = t >> 5;
  int tn    = N >> 7;                       // block tiles along N
  int m0    = ((int)blockIdx.x / tn) << 7;
  int n0    = ((int)blockIdx.x % tn) << 7;
  int wm0   = (wave >> 1) << 6;             // wave offset inside block tile
  int wn0   = (wave & 1) << 6;
  int half  = lane >> 4;
  int colid = lane & 15;

  // Issue the 4 async b128 copies this thread owns for K-slice k0 -> buf.
  auto prefetch = [&](int buf, int k0) {
#pragma unroll
    for (int i = 0; i < 2; ++i) {
      int c  = t + i * 128;                 // chunk id 0..255
      int r  = c >> 1;                      // row 0..127
      int cc = (c & 1) << 3;                // k sub-offset 0 or 8
      async_copy_b128((unsigned)(size_t)&sA[buf][r * LDS_LDK + cc],
                      A + (size_t)(m0 + r) * K + k0 + cc);
      async_copy_b128((unsigned)(size_t)&sB[buf][r * LDS_LDK + cc],
                      W + (size_t)(n0 + r) * K + k0 + cc);
    }
  };

  v8f acc[4][4] = {};

  prefetch(0, 0);
  int nk = K >> 5;
  for (int ki = 0; ki < nk; ++ki) {
    int buf = ki & 1;
    bool more = (ki + 1) < nk;
    if (more) {
      prefetch(buf ^ 1, (ki + 1) << 5);
      // 4 ops outstanding for the *next* buffer; in-order completion means
      // everything for the current buffer has landed in LDS.
      asm volatile("s_wait_asynccnt 0x4" ::: "memory");
    } else {
      asm volatile("s_wait_asynccnt 0x0" ::: "memory");
    }
    __syncthreads();

    v16bf af[4], bfr[4];
#pragma unroll
    for (int i = 0; i < 4; ++i)
      af[i] = load_a_frag(sA[buf], LDS_LDK, wm0 + i * 16, 0, lane);
#pragma unroll
    for (int j = 0; j < 4; ++j)
      bfr[j] = load_b_frag(sB[buf], LDS_LDK, wn0 + j * 16, 0, lane);
#pragma unroll
    for (int i = 0; i < 4; ++i)
#pragma unroll
      for (int j = 0; j < 4; ++j)
        acc[i][j] = WMMA_BF16(af[i], bfr[j], acc[i][j]);

    __syncthreads();
  }

#pragma unroll
  for (int i = 0; i < 4; ++i) {
#pragma unroll
    for (int j = 0; j < 4; ++j) {
      int n = n0 + wn0 + j * 16 + colid;
      float bv = bias[n];
#pragma unroll
      for (int r = 0; r < 8; ++r) {
        int m = m0 + wm0 + i * 16 + half * 8 + r;
        float val = acc[i][j][r] + bv;
        if (mode == 3) {
          ((float*)outp)[(size_t)m * N + n] = val;
        } else if (mode == 2) {
          int hh = n >> 6, d = n & 63;
          ((__bf16*)outp)[((size_t)hh * HDIM + d) * M + m] = (__bf16)val;
        } else {
          int hh = n >> 6, d = n & 63;
          ((__bf16*)outp)[((size_t)hh * M + m) * HDIM + d] = (__bf16)(val * scale);
        }
      }
    }
  }
}

// ---------------------------------------------------------------------------
// Flash attention: 1 wave per (head, 16-row query tile).
// qb: [h][pos][d] bf16 (pre-scaled), kb: [h][pos][d] bf16, vt: [h][d][pos] bf16
// attn_out: [pos][HID] bf16
__global__ __launch_bounds__(32) void attn_wmma(
    const __bf16* __restrict__ qb, const __bf16* __restrict__ kb,
    const __bf16* __restrict__ vt, __bf16* __restrict__ attn_out) {
  __shared__ __align__(16) __bf16 pbuf[16 * 64];

  int lane  = threadIdx.x & 31;
  int q0    = blockIdx.x << 4;
  int h     = blockIdx.y;
  int half  = lane >> 4;
  int colid = lane & 15;

  const __bf16* Qh = qb + (size_t)h * QLEN * HDIM;
  const __bf16* Kh = kb + (size_t)h * QLEN * HDIM;
  const __bf16* Vh = vt + (size_t)h * HDIM * QLEN;

  v16bf aq0 = load_a_frag(Qh, HDIM, q0, 0,  lane);
  v16bf aq1 = load_a_frag(Qh, HDIM, q0, 32, lane);

  float mrun[8], lrun[8];
#pragma unroll
  for (int r = 0; r < 8; ++r) { mrun[r] = -INFINITY; lrun[r] = 0.f; }
  v8f acc[4] = {};

  int kend = q0 + 16;  // number of keys this tile attends to
  for (int kbase = 0; kbase < kend; kbase += 64) {
    // ---- S = Q K^T for 16 rows x 64 keys (4 sub-tiles) -------------------
    v8f sv[4];
#pragma unroll
    for (int sub = 0; sub < 4; ++sub) {
      int pos0 = kbase + sub * 16;
      v16bf b0 = load_b_frag(Kh, HDIM, pos0, 0,  lane);
      v16bf b1 = load_b_frag(Kh, HDIM, pos0, 32, lane);
      v8f c = {};
      c = WMMA_BF16(aq0, b0, c);
      c = WMMA_BF16(aq1, b1, c);
      int key = pos0 + colid;
#pragma unroll
      for (int r = 0; r < 8; ++r) {
        int qrow = q0 + half * 8 + r;
        sv[sub][r] = (key <= qrow) ? c[r] : -1e30f;
      }
    }

    // ---- online softmax --------------------------------------------------
    float mnew[8], alpha[8];
#pragma unroll
    for (int r = 0; r < 8; ++r) {
      float tmx = fmaxf(fmaxf(sv[0][r], sv[1][r]), fmaxf(sv[2][r], sv[3][r]));
      tmx = fmaxf(tmx, __shfl_xor(tmx, 1, 32));
      tmx = fmaxf(tmx, __shfl_xor(tmx, 2, 32));
      tmx = fmaxf(tmx, __shfl_xor(tmx, 4, 32));
      tmx = fmaxf(tmx, __shfl_xor(tmx, 8, 32));
      mnew[r]  = fmaxf(mrun[r], tmx);
      alpha[r] = __expf(mrun[r] - mnew[r]);
      mrun[r]  = mnew[r];
    }
#pragma unroll
    for (int n = 0; n < 4; ++n)
#pragma unroll
      for (int r = 0; r < 8; ++r) acc[n][r] *= alpha[r];

    float psum[8];
#pragma unroll
    for (int r = 0; r < 8; ++r) psum[r] = 0.f;
#pragma unroll
    for (int sub = 0; sub < 4; ++sub) {
#pragma unroll
      for (int r = 0; r < 8; ++r) {
        float p = __expf(sv[sub][r] - mnew[r]);
        psum[r] += p;
        pbuf[(half * 8 + r) * 64 + sub * 16 + colid] = (__bf16)p;
      }
    }
#pragma unroll
    for (int r = 0; r < 8; ++r) {
      float tsm = psum[r];
      tsm += __shfl_xor(tsm, 1, 32);
      tsm += __shfl_xor(tsm, 2, 32);
      tsm += __shfl_xor(tsm, 4, 32);
      tsm += __shfl_xor(tsm, 8, 32);
      lrun[r] = lrun[r] * alpha[r] + tsm;
    }

    __syncthreads();  // single-wave WG: orders LDS store->load for compiler
    v16bf pa0 = load_a_frag(pbuf, 64, 0, 0,  lane);
    v16bf pa1 = load_a_frag(pbuf, 64, 0, 32, lane);
    __syncthreads();

    // ---- O += P V --------------------------------------------------------
#pragma unroll
    for (int n = 0; n < 4; ++n) {
      v16bf vb0 = load_b_frag(Vh, QLEN, n * 16, kbase,      lane);
      v16bf vb1 = load_b_frag(Vh, QLEN, n * 16, kbase + 32, lane);
      acc[n] = WMMA_BF16(pa0, vb0, acc[n]);
      acc[n] = WMMA_BF16(pa1, vb1, acc[n]);
    }
  }

  // ---- normalize + store [pos][HID] bf16 ---------------------------------
#pragma unroll
  for (int n = 0; n < 4; ++n) {
#pragma unroll
    for (int r = 0; r < 8; ++r) {
      int m = q0 + half * 8 + r;
      int d = n * 16 + colid;
      float o = acc[n][r] / lrun[r];
      attn_out[(size_t)m * HID + h * HDIM + d] = (__bf16)o;
    }
  }
}

// ---------------------------------------------------------------------------
extern "C" void kernel_launch(void* const* d_in, const int* in_sizes, int n_in,
                              void* d_out, int out_size, void* d_ws, size_t ws_size,
                              hipStream_t stream) {
  const float* hidden = (const float*)d_in[0];
  // d_in[1] position_ids, d_in[2] mask, d_in[3]/[4] kv caches: not needed
  const float* q_w = (const float*)d_in[5];
  const float* q_b = (const float*)d_in[6];
  const float* k_w = (const float*)d_in[7];
  const float* k_b = (const float*)d_in[8];
  const float* v_w = (const float*)d_in[9];
  const float* v_b = (const float*)d_in[10];
  const float* o_w = (const float*)d_in[11];
  const float* o_b = (const float*)d_in[12];
  float* out = (float*)d_out;

  char* ws = (char*)d_ws;
  size_t off = 0;
  const size_t MAT = (size_t)QLEN * HID;  // elements per 2048x2048 matrix
  __bf16* xb   = (__bf16*)(ws + off); off += MAT * 2;
  __bf16* wq   = (__bf16*)(ws + off); off += MAT * 2;
  __bf16* wk   = (__bf16*)(ws + off); off += MAT * 2;
  __bf16* wv   = (__bf16*)(ws + off); off += MAT * 2;
  __bf16* wo   = (__bf16*)(ws + off); off += MAT * 2;
  __bf16* qbuf = (__bf16*)(ws + off); off += MAT * 2;   // [h][pos][d], pre-scaled
  __bf16* kbuf = (__bf16*)(ws + off); off += MAT * 2;   // [h][pos][d]
  __bf16* vtb  = (__bf16*)(ws + off); off += MAT * 2;   // [h][d][pos]
  __bf16* attn = (__bf16*)(ws + off); off += MAT * 2;   // [pos][HID]

  // 1) f32 -> bf16 conversions
  cvt_f32_bf16<<<2048, 256, 0, stream>>>(hidden, xb, (int)MAT);
  cvt_f32_bf16<<<2048, 256, 0, stream>>>(q_w, wq, (int)MAT);
  cvt_f32_bf16<<<2048, 256, 0, stream>>>(k_w, wk, (int)MAT);
  cvt_f32_bf16<<<2048, 256, 0, stream>>>(v_w, wv, (int)MAT);
  cvt_f32_bf16<<<2048, 256, 0, stream>>>(o_w, wo, (int)MAT);

  // 2) QKV projections: 16x16 grid of 128x128 block tiles
  gemm_wmma<<<256, 128, 0, stream>>>(xb, wq, q_b, qbuf, 0, ATT_SCALE, QLEN, HID, HID);
  gemm_wmma<<<256, 128, 0, stream>>>(xb, wk, k_b, kbuf, 0, 1.0f,      QLEN, HID, HID);
  gemm_wmma<<<256, 128, 0, stream>>>(xb, wv, v_b, vtb,  2, 1.0f,      QLEN, HID, HID);

  // 3) causal flash attention: one wave per (16 queries, head)
  attn_wmma<<<dim3(QLEN / 16, NHEAD), 32, 0, stream>>>(qbuf, kbuf, vtb, attn);

  // 4) output projection -> f32 d_out
  gemm_wmma<<<256, 128, 0, stream>>>(attn, wo, o_b, out, 3, 1.0f, QLEN, HID, HID);
}
